// LSTMModel_49400713838612
// MI455X (gfx1250) — matrix operand order repaired and checked by reading, not compiled, not run on hardware
//
#include <hip/hip_runtime.h>
#include <hip/hip_fp16.h>

typedef __attribute__((ext_vector_type(16))) _Float16 v16h;
typedef __attribute__((ext_vector_type(8)))  _Float16 v8h;
typedef __attribute__((ext_vector_type(8)))  float    v8f;

#define B_  64
#define T_  512
#define I_  64
#define H_  512
#define G4H 2048

// CDNA5 async global->LDS copy path (ASYNCcnt), if this toolchain exposes it.
#if defined(__has_builtin)
#  if __has_builtin(__builtin_amdgcn_global_load_async_to_lds_b128) && \
      __has_builtin(__builtin_amdgcn_s_wait_asynccnt)
#    define USE_ASYNC_LDS 1
#  endif
#endif

#ifdef USE_ASYNC_LDS
typedef int v4i_ __attribute__((vector_size(16)));
#  define GLB_P(p) ((__attribute__((address_space(1))) v4i_*)(p))
#  define LDS_P(p) ((__attribute__((address_space(3))) v4i_*)(p))
#endif

// ---------------------------------------------------------------- utilities

__global__ void f32_to_f16_kernel(const float* __restrict__ src,
                                  _Float16* __restrict__ dst, int n) {
    int i = blockIdx.x * blockDim.x + threadIdx.x;
    if (i < n) dst[i] = (_Float16)src[i];
}

__global__ void bias_sum_kernel(const float* __restrict__ bih0, const float* __restrict__ bhh0,
                                const float* __restrict__ bih1, const float* __restrict__ bhh1,
                                const float* __restrict__ bih2, const float* __restrict__ bhh2,
                                float* __restrict__ bsum) {
    int i = blockIdx.x * blockDim.x + threadIdx.x;
    if (i >= 3 * G4H) return;
    int l = i >> 11, j = i & (G4H - 1);
    const float* a = (l == 0) ? bih0 : (l == 1) ? bih1 : bih2;
    const float* b = (l == 0) ? bhh0 : (l == 1) ? bhh1 : bhh2;
    bsum[i] = a[j] + b[j];
}

// Generation-count grid barrier (32 co-resident blocks).
__device__ __forceinline__ void grid_barrier(unsigned* cnt, unsigned* gen, unsigned nblk) {
    __syncthreads();
    if (threadIdx.x == 0) {
        __threadfence();  // release h stores to device scope
        unsigned g = __hip_atomic_load(gen, __ATOMIC_RELAXED, __HIP_MEMORY_SCOPE_AGENT);
        if (atomicAdd(cnt, 1u) == nblk - 1) {
            __hip_atomic_store(cnt, 0u, __ATOMIC_RELAXED, __HIP_MEMORY_SCOPE_AGENT);
            __threadfence();
            atomicAdd(gen, 1u);
        } else {
            while (__hip_atomic_load(gen, __ATOMIC_RELAXED, __HIP_MEMORY_SCOPE_AGENT) == g)
                __builtin_amdgcn_s_sleep(2);
        }
        __threadfence();  // acquire side
    }
    __syncthreads();
}

// ------------------------------------------------- persistent LSTM layer
// grid = 32 blocks (one 16-wide h-column tile each), block = 512 thr = 16 waves.
// wave (mTile = wv&3, gate = wv>>2) owns one 16x16 WMMA output tile per step.
template <int DIN>
__global__ void lstm_layer_kernel(const _Float16* __restrict__ in,   // [B,T,DIN]
                                  const _Float16* __restrict__ wih,  // [4H,DIN]
                                  const _Float16* __restrict__ whh,  // [4H,H]
                                  const float*    __restrict__ bsum, // [4H]
                                  _Float16* __restrict__ out,        // [B,T,H]
                                  _Float16* __restrict__ hbuf,       // [B,H]
                                  unsigned* __restrict__ bar) {
    constexpr int DTOT   = DIN + H_;
    constexpr int STRIDE = DTOT + 8;        // +16B pad: stagger LDS banks
    extern __shared__ char smem[];
    _Float16* xh = (_Float16*)smem;                  // [64][STRIDE]  A: [x_t | h]
    _Float16* wl = xh + 64 * STRIDE;                 // [64][STRIDE]  B: weight cols
    float*    gl = (float*)(wl + 64 * STRIDE);       // [4*64][17]    gate tiles
    float*    cl = gl + 4 * 64 * 17;                 // [64][17]      cell state

    const int tid   = threadIdx.x;
    const int lane  = tid & 31;
    const int wv    = tid >> 5;
    const int mTile = wv & 3;
    const int gate  = wv >> 2;
    const int col0  = blockIdx.x * 16;

    // one-time: stage this WG's weight columns (concat Wih|Whh rows) into LDS
    for (int i = tid; i < 64 * DTOT; i += 512) {
        int c = i / DTOT, k = i - c * DTOT;
        int n = (c >> 4) * H_ + col0 + (c & 15);     // global gate-row
        wl[c * STRIDE + k] = (k < DIN) ? wih[(size_t)n * DIN + k]
                                       : whh[(size_t)n * H_ + (k - DIN)];
    }
    for (int i = tid; i < 64 * 17; i += 512) cl[i] = 0.0f;

    const float bv = bsum[gate * H_ + col0 + (lane & 15)];  // per-column bias
    const _Float16* arow = xh + (mTile * 16 + (lane & 15)) * STRIDE;
    const _Float16* brow = wl + (gate * 16 + (lane & 15)) * STRIDE;
    const int ka = (lane & 16) ? 8 : 0;    // A frag: K halves per ISA layout
    const int kb = (lane & 16) ? 8 : 0;    // B frag: K halves per ISA layout
    __syncthreads();

    for (int t = 0; t < T_; ++t) {
        // ---- stage x_t and h_{t-1} into LDS ----
        constexpr int XV = DIN / 8;   // uint4 per A row for x slice
#ifdef USE_ASYNC_LDS
        for (int i = tid; i < 64 * XV; i += 512) {
            int b = i / XV, q = i - b * XV;
            __builtin_amdgcn_global_load_async_to_lds_b128(
                GLB_P(in + ((size_t)b * T_ + t) * DIN + q * 8),
                LDS_P(xh + b * STRIDE + q * 8), 0, 0);
        }
        if (t == 0) {
            for (int i = tid; i < 64 * 64; i += 512) {
                int b = i >> 6, q = i & 63;
                ((uint4*)(xh + b * STRIDE + DIN))[q] = make_uint4(0, 0, 0, 0);
            }
        } else {
            for (int i = tid; i < 64 * 64; i += 512) {
                int b = i >> 6, q = i & 63;
                __builtin_amdgcn_global_load_async_to_lds_b128(
                    GLB_P(hbuf + b * H_ + q * 8),
                    LDS_P(xh + b * STRIDE + DIN + q * 8), 0, 0);
            }
        }
        __builtin_amdgcn_s_wait_asynccnt(0);
#else
        for (int i = tid; i < 64 * XV; i += 512) {
            int b = i / XV, q = i - b * XV;
            ((uint4*)(xh + b * STRIDE))[q] =
                ((const uint4*)(in + ((size_t)b * T_ + t) * DIN))[q];
        }
        for (int i = tid; i < 64 * 64; i += 512) {
            int b = i >> 6, q = i & 63;
            uint4 v = (t == 0) ? make_uint4(0, 0, 0, 0)
                               : ((const uint4*)(hbuf + b * H_))[q];
            ((uint4*)(xh + b * STRIDE + DIN))[q] = v;
        }
#endif
        __syncthreads();

        // ---- WMMA K-loop over [x_t | h] ----
        v8f acc = {bv, bv, bv, bv, bv, bv, bv, bv};
#pragma unroll 4
        for (int k = 0; k < DTOT; k += 32) {
            v8h a0 = *(const v8h*)(arow + k + ka);
            v8h a1 = *(const v8h*)(arow + k + 16 + ka);
            v16h a = __builtin_shufflevector(a0, a1, 0,1,2,3,4,5,6,7,8,9,10,11,12,13,14,15);
            v8h b0 = *(const v8h*)(brow + k + kb);
            v8h b1 = *(const v8h*)(brow + k + 16 + kb);
            v16h b = __builtin_shufflevector(b0, b1, 0,1,2,3,4,5,6,7,8,9,10,11,12,13,14,15);
            acc = __builtin_amdgcn_wmma_f32_16x16x32_f16(false, a, false, b,
                                                         (short)0, acc, false, false);
        }

        // scatter gate tile to LDS (D layout: VGPR r -> row (lane<16 ? r : 8+r))
        {
            int rbase = mTile * 16 + ((lane & 16) ? 8 : 0);
            float* gp = gl + ((gate * 64 + rbase) * 17 + (lane & 15));
#pragma unroll
            for (int r = 0; r < 8; ++r) gp[r * 17] = acc[r];
        }
        __syncthreads();

        // ---- pointwise LSTM cell for this WG's 64x16 slice ----
        for (int e = tid; e < 1024; e += 512) {
            int m = e >> 4, jl = e & 15;
            float ig = gl[(0 * 64 + m) * 17 + jl];
            float fg = gl[(1 * 64 + m) * 17 + jl];
            float gg = gl[(2 * 64 + m) * 17 + jl];
            float og = gl[(3 * 64 + m) * 17 + jl];
            ig = 1.0f / (1.0f + __expf(-ig));
            fg = 1.0f / (1.0f + __expf(-fg));
            og = 1.0f / (1.0f + __expf(-og));
            gg = tanhf(gg);
            float c = fg * cl[m * 17 + jl] + ig * gg;
            cl[m * 17 + jl] = c;
            _Float16 hh = (_Float16)(og * tanhf(c));
            hbuf[m * H_ + col0 + jl] = hh;
            out[((size_t)m * T_ + t) * H_ + col0 + jl] = hh;
        }
        grid_barrier(bar, bar + 1, gridDim.x);
    }
}

// ------------------------------------------------------------------- FC head
__global__ void fc_kernel(const _Float16* __restrict__ traj, const float* __restrict__ fcw,
                          const float* __restrict__ fcb, float* __restrict__ outp) {
    __shared__ float red[256];
    int b = blockIdx.x;
    const _Float16* h = traj + ((size_t)b * T_ + (T_ - 1)) * H_;
    float s = 0.f;
    for (int j = threadIdx.x; j < H_; j += 256) s += (float)h[j] * fcw[j];
    red[threadIdx.x] = s;
    __syncthreads();
    for (int off = 128; off > 0; off >>= 1) {
        if (threadIdx.x < off) red[threadIdx.x] += red[threadIdx.x + off];
        __syncthreads();
    }
    if (threadIdx.x == 0) outp[b] = red[0] + fcb[0];
}

// ---------------------------------------------------------------------- host

static constexpr int lds_bytes(int DIN) {
    int STRIDE = DIN + H_ + 8;
    return 64 * STRIDE * 2 * 2 + (4 * 64 * 17 + 64 * 17) * 4;
}

extern "C" void kernel_launch(void* const* d_in, const int* in_sizes, int n_in,
                              void* d_out, int out_size, void* d_ws, size_t ws_size,
                              hipStream_t stream) {
    const float* x = (const float*)d_in[0];
    const float* wih[3] = {(const float*)d_in[1], (const float*)d_in[5], (const float*)d_in[9]};
    const float* whh[3] = {(const float*)d_in[2], (const float*)d_in[6], (const float*)d_in[10]};
    const float* bih[3] = {(const float*)d_in[3], (const float*)d_in[7], (const float*)d_in[11]};
    const float* bhh[3] = {(const float*)d_in[4], (const float*)d_in[8], (const float*)d_in[12]};
    const float* fcw = (const float*)d_in[13];
    const float* fcb = (const float*)d_in[14];
    float* outp = (float*)d_out;

    char* ws = (char*)d_ws;
    size_t off = 0;
    auto alloc = [&](size_t bytes) -> void* {
        void* p = ws + off;
        off = (off + bytes + 255) & ~(size_t)255;
        return p;
    };
    _Float16* x16   = (_Float16*)alloc((size_t)B_ * T_ * I_ * 2);
    _Float16* traj1 = (_Float16*)alloc((size_t)B_ * T_ * H_ * 2);
    _Float16* traj2 = (_Float16*)alloc((size_t)B_ * T_ * H_ * 2);
    const int wsz[6] = {G4H * I_, G4H * H_, G4H * H_, G4H * H_, G4H * H_, G4H * H_};
    _Float16* w16[6];
    for (int i = 0; i < 6; ++i) w16[i] = (_Float16*)alloc((size_t)wsz[i] * 2);
    float*    bsum = (float*)alloc(3 * G4H * 4);
    _Float16* hbuf = (_Float16*)alloc((size_t)B_ * H_ * 2);
    unsigned* bar  = (unsigned*)alloc(256);

    (void)hipMemsetAsync(bar, 0, 256, stream);

    const float* wsrc[6] = {wih[0], whh[0], wih[1], whh[1], wih[2], whh[2]};
    {
        int n = B_ * T_ * I_;
        f32_to_f16_kernel<<<(n + 255) / 256, 256, 0, stream>>>(x, x16, n);
        for (int i = 0; i < 6; ++i)
            f32_to_f16_kernel<<<(wsz[i] + 255) / 256, 256, 0, stream>>>(wsrc[i], w16[i], wsz[i]);
        bias_sum_kernel<<<(3 * G4H + 255) / 256, 256, 0, stream>>>(
            bih[0], bhh[0], bih[1], bhh[1], bih[2], bhh[2], bsum);
    }

    constexpr int lds0 = lds_bytes(I_);   // layer 0
    constexpr int lds1 = lds_bytes(H_);   // layers 1,2
    (void)hipFuncSetAttribute(reinterpret_cast<const void*>(&lstm_layer_kernel<I_>),
                              hipFuncAttributeMaxDynamicSharedMemorySize, lds0);
    (void)hipFuncSetAttribute(reinterpret_cast<const void*>(&lstm_layer_kernel<H_>),
                              hipFuncAttributeMaxDynamicSharedMemorySize, lds1);

    lstm_layer_kernel<I_><<<32, 512, lds0, stream>>>(x16,   w16[0], w16[1], bsum + 0 * G4H, traj1, hbuf, bar);
    lstm_layer_kernel<H_><<<32, 512, lds1, stream>>>(traj1, w16[2], w16[3], bsum + 1 * G4H, traj2, hbuf, bar);
    lstm_layer_kernel<H_><<<32, 512, lds1, stream>>>(traj2, w16[4], w16[5], bsum + 2 * G4H, traj1, hbuf, bar);

    fc_kernel<<<64, 256, 0, stream>>>(traj1, fcw, fcb, outp);
}
